// GCNBiLSTM_53901839565300
// MI455X (gfx1250) — matrix operand order repaired and checked by reading, not compiled, not run on hardware
//
#include <hip/hip_runtime.h>
#include <math.h>

#define NN 100000
#define EE 1600000

typedef __attribute__((ext_vector_type(16))) __bf16 v16bf;
typedef __attribute__((ext_vector_type(8)))  float  v8f;
typedef unsigned int u32x4 __attribute__((ext_vector_type(4)));
typedef int          i32x4 __attribute__((ext_vector_type(4)));
typedef int          i32x8 __attribute__((ext_vector_type(8)));

union FragU { uint4 q[2]; v16bf b; };

__device__ __forceinline__ unsigned short f2bf(float x) {
    union { float f; unsigned int u; } v; v.f = x;
    unsigned int u = v.u;
    u += 0x7FFFu + ((u >> 16) & 1u);           // round-to-nearest-even
    return (unsigned short)(u >> 16);
}

// A fragment (16x32 bf16, row-major source, rowp already points at this lane's row).
// ISA 7.12.2: lanes0-15 elems0-7 = K0..7, elems8-15 = K16..23; lanes16-31 shifted +8.
__device__ __forceinline__ v16bf load_a16(const unsigned short* rowp, int kBase, int lane) {
    int off = kBase + ((lane & 16) ? 8 : 0);
    FragU f;
    f.q[0] = *reinterpret_cast<const uint4*>(rowp + off);
    f.q[1] = *reinterpret_cast<const uint4*>(rowp + off + 16);
    return f.b;
}

// B fragment (32x16 bf16) from Bt layout [Ncol][K]; colp points at this lane's column row.
// lanes0-15 elems0-7 = K0..7, elems8-15 = K16..23; lanes16-31 shifted +8 (matches A).
__device__ __forceinline__ v16bf load_b16(const unsigned short* colp, int kBase, int lane) {
    int off = kBase + ((lane & 16) ? 8 : 0);
    FragU f;
    f.q[0] = *reinterpret_cast<const uint4*>(colp + off);
    f.q[1] = *reinterpret_cast<const uint4*>(colp + off + 16);
    return f.b;
}

// TDM: DMA `elems` bf16 elements (1-D tile) from global `ga` into LDS offset `ldsOff`.
// D# layout per CDNA5 ISA sect.8. EXEC is ignored; issue from one wave.
__device__ __forceinline__ void tdm_load_1d(unsigned long long ga, unsigned int ldsOff,
                                            unsigned int elems) {
    u32x4 g0 = { 1u,                                   // count=1, user mode
                 ldsOff,                               // lds_addr
                 (unsigned int)ga,                     // global_addr[31:0]
                 (unsigned int)((ga >> 32) & 0x01FFFFFFull) | 0x80000000u }; // ga[56:32] | type=2
    i32x8 g1 = { (int)0x00010000u,                     // data_size=1 (2 bytes)
                 (int)((elems & 0xFFFFu) << 16),       // tensor_dim0[15:0]
                 (int)((elems >> 16) | (1u << 16)),    // tensor_dim0[31:16] | tensor_dim1=1
                 (int)((elems & 0xFFFFu) << 16),       // tile_dim0 = elems (<=65535)
                 1,                                    // tile_dim1 = 1
                 (int)elems,                           // tensor_dim0_stride[31:0]
                 0, 0 };
    i32x4 gz = { 0, 0, 0, 0 };
#if __clang_major__ >= 23
    i32x8 gz8 = { 0, 0, 0, 0, 0, 0, 0, 0 };
    __builtin_amdgcn_tensor_load_to_lds(g0, g1, gz, gz, gz8, 0);
#else
    __builtin_amdgcn_tensor_load_to_lds(g0, g1, gz, gz, 0);
#endif
}

// ---------------- small prep kernels ----------------
__global__ void k_deg_init(float* deg, int n) {
    int i = blockIdx.x * blockDim.x + threadIdx.x;
    if (i < n) deg[i] = 1.0f;                  // self-loop
}
__global__ void k_deg(const int* __restrict__ dst, float* deg, int e) {
    int i = blockIdx.x * blockDim.x + threadIdx.x;
    if (i < e) atomicAdd(&deg[dst[i]], 1.0f);
}
__global__ void k_rsqrt(const float* __restrict__ deg, float* dinv, int n) {
    int i = blockIdx.x * blockDim.x + threadIdx.x;
    if (i < n) dinv[i] = rsqrtf(fmaxf(deg[i], 1e-12f));
}
__global__ void k_f2bf(const float* __restrict__ src, unsigned short* __restrict__ dst, int n) {
    int i = blockIdx.x * blockDim.x + threadIdx.x;
    if (i < n) dst[i] = f2bf(src[i]);
}
// dst[nn][kk] = src[kk][nn], src is K x NC
__global__ void k_t2bf(const float* __restrict__ src, unsigned short* __restrict__ dst, int K, int NC) {
    int i = blockIdx.x * blockDim.x + threadIdx.x;
    if (i >= K * NC) return;
    int nn = i / K, kk = i % K;
    dst[i] = f2bf(src[kk * NC + nn]);
}

// ---------------- bf16 WMMA GEMM: C[M x128] = A_bf[M x128] @ Bt_bf[128col][128K] ----------------
// Weight matrix (32KB) is DMA'd into LDS by the Tensor Data Mover, then all 8 waves feed
// their B fragments from LDS (ds_load_b128) while A streams from global.
__global__ __launch_bounds__(256) void k_gemm128(const unsigned short* __restrict__ A,
                                                 const unsigned short* __restrict__ Bt,
                                                 float* __restrict__ C, int mtiles) {
    __shared__ unsigned short sB[128 * 128];
    int lane = threadIdx.x & 31;
    int wave = threadIdx.x >> 5;

    if (threadIdx.x == 0) {                        // TDM ignores EXEC: one issue from wave 0
        tdm_load_1d((unsigned long long)(uintptr_t)Bt,
                    (unsigned int)(uintptr_t)(void*)sB, 128u * 128u);
    }
    if (threadIdx.x < 32) __builtin_amdgcn_s_wait_tensorcnt(0);
    __syncthreads();

    int mt = blockIdx.x * 8 + wave;
    if (mt >= mtiles) return;
    const unsigned short* rowp = A + (size_t)(mt * 16 + (lane & 15)) * 128;
    int cl = lane & 15;
    v8f z = {};
    v8f acc[8];
#pragma unroll
    for (int i = 0; i < 8; i++) acc[i] = z;
#pragma unroll
    for (int ks = 0; ks < 4; ks++) {
        v16bf a = load_a16(rowp, ks * 32, lane);
        v16bf b[8];
#pragma unroll
        for (int nt = 0; nt < 8; nt++)
            b[nt] = load_b16(sB + (size_t)(nt * 16 + cl) * 128, ks * 32, lane);
#pragma unroll
        for (int nt = 0; nt < 8; nt++)
            acc[nt] = __builtin_amdgcn_wmma_f32_16x16x32_bf16(false, a, false, b[nt],
                                                              (short)0, acc[nt], false, false);
    }
    int rbase = mt * 16 + ((lane & 16) ? 8 : 0);
#pragma unroll
    for (int nt = 0; nt < 8; nt++)
#pragma unroll
        for (int r = 0; r < 8; r++)
            C[(size_t)(rbase + r) * 128 + nt * 16 + cl] = acc[nt][r];
}

// ---------------- GCN aggregation ----------------
__global__ void k_selfloop(const float* __restrict__ xl, const float* __restrict__ dinv,
                           float* __restrict__ agg, int n) {
    int i = blockIdx.x * blockDim.x + threadIdx.x;   // n = N*128
    if (i >= n) return;
    float w = dinv[i >> 7];
    agg[i] = xl[i] * w * w;
}
__global__ __launch_bounds__(256) void k_scatter(const int* __restrict__ src, const int* __restrict__ dst,
                                                 const float* __restrict__ dinv, const float* __restrict__ xl,
                                                 float* __restrict__ agg, int e) {
    int gid = blockIdx.x * blockDim.x + threadIdx.x;
    int eidx = gid >> 5;
    if (eidx >= e) return;
    int c = (gid & 31) * 4;
    int s = src[eidx], d = dst[eidx];
    float norm = dinv[s] * dinv[d];
    float4 v = *reinterpret_cast<const float4*>(xl + (size_t)s * 128 + c);
    float* o = agg + (size_t)d * 128 + c;
    atomicAdd(o + 0, v.x * norm);
    atomicAdd(o + 1, v.y * norm);
    atomicAdd(o + 2, v.z * norm);
    atomicAdd(o + 3, v.w * norm);
}
__global__ void k_bias_relu_bf(const float* __restrict__ agg, const float* __restrict__ b,
                               unsigned short* __restrict__ out, int n) {
    int i = blockIdx.x * blockDim.x + threadIdx.x;
    if (i >= n) return;
    float v = agg[i] + b[i & 127];
    out[i] = f2bf(v > 0.f ? v : 0.f);
}

// ---------------- fused BiLSTM (1 step, h0=c0=0) + FC1 + FC2 ----------------
__global__ __launch_bounds__(128) void k_lstm_fc(const unsigned short* __restrict__ H,
    const unsigned short* __restrict__ WihF, const unsigned short* __restrict__ WihB,
    const float* __restrict__ bihF, const float* __restrict__ bhhF,
    const float* __restrict__ bihB, const float* __restrict__ bhhB,
    const unsigned short* __restrict__ fcW1t, const float* __restrict__ fcb1,
    const float* __restrict__ fcW2, const float* __restrict__ fcb2,
    float* __restrict__ out, int mtiles) {
    __shared__ unsigned short sBi[4][16 * 256];   // [hf | hb] tile per wave, bf16
    __shared__ float sY1[4][16 * 64];             // relu(fc1) tile per wave
    int lane = threadIdx.x & 31;
    int wave = threadIdx.x >> 5;
    int mt = blockIdx.x * 4 + wave;
    if (mt >= mtiles) return;
    int cl = lane & 15;
    int rhi = (lane & 16) ? 8 : 0;
    v8f z = {};

    const unsigned short* rowp = H + (size_t)(mt * 16 + cl) * 128;
    v16bf a[4];
#pragma unroll
    for (int ks = 0; ks < 4; ks++) a[ks] = load_a16(rowp, ks * 32, lane);

    for (int dir = 0; dir < 2; dir++) {
        const unsigned short* Wih = dir ? WihB : WihF;
        const float* bih = dir ? bihB : bihF;
        const float* bhh = dir ? bhhB : bhhF;
#pragma unroll
        for (int nc = 0; nc < 8; nc++) {
            // batch all 12 B fragments, then burst the WMMAs
            v16bf bI[4], bG[4], bO[4];
#pragma unroll
            for (int ks = 0; ks < 4; ks++) {
                bI[ks] = load_b16(Wih + (size_t)(0   + nc * 16 + cl) * 128, ks * 32, lane);
                bG[ks] = load_b16(Wih + (size_t)(256 + nc * 16 + cl) * 128, ks * 32, lane);
                bO[ks] = load_b16(Wih + (size_t)(384 + nc * 16 + cl) * 128, ks * 32, lane);
            }
            v8f ai = z, ag = z, ao = z;        // gates i, g, o (f gate dead: c0 = 0)
#pragma unroll
            for (int ks = 0; ks < 4; ks++) {
                ai = __builtin_amdgcn_wmma_f32_16x16x32_bf16(false, a[ks], false, bI[ks], (short)0, ai, false, false);
                ag = __builtin_amdgcn_wmma_f32_16x16x32_bf16(false, a[ks], false, bG[ks], (short)0, ag, false, false);
                ao = __builtin_amdgcn_wmma_f32_16x16x32_bf16(false, a[ks], false, bO[ks], (short)0, ao, false, false);
            }
            int gc = nc * 16 + cl;
            float bI_ = bih[gc]       + bhh[gc];
            float bG_ = bih[256 + gc] + bhh[256 + gc];
            float bO_ = bih[384 + gc] + bhh[384 + gc];
#pragma unroll
            for (int r = 0; r < 8; r++) {
                float iv = 1.f / (1.f + __expf(-(ai[r] + bI_)));
                float gv = tanhf(ag[r] + bG_);
                float ov = 1.f / (1.f + __expf(-(ao[r] + bO_)));
                float hv = ov * tanhf(iv * gv);
                sBi[wave][(r + rhi) * 256 + dir * 128 + gc] = f2bf(hv);
            }
        }
    }
    asm volatile("s_wait_dscnt 0x0" ::: "memory");   // wave-local LDS visibility

    // FC1: [16x256] @ Bt[64col][256K]
    v8f y[4];
#pragma unroll
    for (int i = 0; i < 4; i++) y[i] = z;
    const unsigned short* srow = &sBi[wave][cl * 256];
#pragma unroll
    for (int ks = 0; ks < 8; ks++) {
        v16bf af = load_a16(srow, ks * 32, lane);
        v16bf bb[4];
#pragma unroll
        for (int nt = 0; nt < 4; nt++)
            bb[nt] = load_b16(fcW1t + (size_t)(nt * 16 + cl) * 256, ks * 32, lane);
#pragma unroll
        for (int nt = 0; nt < 4; nt++)
            y[nt] = __builtin_amdgcn_wmma_f32_16x16x32_bf16(false, af, false, bb[nt], (short)0, y[nt], false, false);
    }
#pragma unroll
    for (int nt = 0; nt < 4; nt++) {
        float bb = fcb1[nt * 16 + cl];
#pragma unroll
        for (int r = 0; r < 8; r++) {
            float v = y[nt][r] + bb;
            sY1[wave][(r + rhi) * 64 + nt * 16 + cl] = v > 0.f ? v : 0.f;
        }
    }
    asm volatile("s_wait_dscnt 0x0" ::: "memory");

    // FC2: 64 -> 1
    if (lane < 16) {
        float acc = fcb2[0];
#pragma unroll
        for (int c = 0; c < 64; c++) acc += sY1[wave][lane * 64 + c] * fcW2[c];
        out[mt * 16 + lane] = acc;
    }
}

extern "C" void kernel_launch(void* const* d_in, const int* in_sizes, int n_in,
                              void* d_out, int out_size, void* d_ws, size_t ws_size,
                              hipStream_t stream) {
    (void)in_sizes; (void)n_in; (void)out_size; (void)ws_size;
    const float* x    = (const float*)d_in[0];
    const int*   ei   = (const int*)d_in[1];
    const int*   src  = ei;
    const int*   dstp = ei + EE;
    const float* W1   = (const float*)d_in[2];
    const float* b1   = (const float*)d_in[3];
    const float* W2   = (const float*)d_in[4];
    const float* b2   = (const float*)d_in[5];
    const float* WihF = (const float*)d_in[6];
    const float* bihF = (const float*)d_in[8];
    const float* bhhF = (const float*)d_in[9];
    const float* WihB = (const float*)d_in[10];
    const float* bihB = (const float*)d_in[12];
    const float* bhhB = (const float*)d_in[13];
    const float* fcW1 = (const float*)d_in[14];
    const float* fcb1 = (const float*)d_in[15];
    const float* fcW2 = (const float*)d_in[16];
    const float* fcb2 = (const float*)d_in[17];
    float* out = (float*)d_out;

    char* w = (char*)d_ws;
    auto carve = [&](size_t bytes) -> char* {
        char* p = w; w += (bytes + 255) & ~(size_t)255; return p;
    };
    float*          deg    = (float*)carve((size_t)NN * 4);
    float*          dinv   = (float*)carve((size_t)NN * 4);
    unsigned short* bufBF  = (unsigned short*)carve((size_t)NN * 128 * 2); // x_bf -> h1_bf -> h2_bf
    float*          xl     = (float*)carve((size_t)NN * 128 * 4);
    float*          agg    = (float*)carve((size_t)NN * 128 * 4);
    unsigned short* W1t    = (unsigned short*)carve(128 * 128 * 2);
    unsigned short* W2t    = (unsigned short*)carve(128 * 128 * 2);
    unsigned short* WihFb  = (unsigned short*)carve(512 * 128 * 2);
    unsigned short* WihBb  = (unsigned short*)carve(512 * 128 * 2);
    unsigned short* fcW1t  = (unsigned short*)carve(64 * 256 * 2);

    const int NF = NN * 128;
    const int mtiles = NN / 16;                       // 6250 exactly

    // norms
    k_deg_init<<<(NN + 255) / 256, 256, 0, stream>>>(deg, NN);
    k_deg<<<(EE + 255) / 256, 256, 0, stream>>>(dstp, deg, EE);
    k_rsqrt<<<(NN + 255) / 256, 256, 0, stream>>>(deg, dinv, NN);

    // weight / activation conversions
    k_f2bf<<<(NF + 255) / 256, 256, 0, stream>>>(x, bufBF, NF);
    k_t2bf<<<(128 * 128 + 255) / 256, 256, 0, stream>>>(W1, W1t, 128, 128);
    k_t2bf<<<(128 * 128 + 255) / 256, 256, 0, stream>>>(W2, W2t, 128, 128);
    k_f2bf<<<(512 * 128 + 255) / 256, 256, 0, stream>>>(WihF, WihFb, 512 * 128);
    k_f2bf<<<(512 * 128 + 255) / 256, 256, 0, stream>>>(WihB, WihBb, 512 * 128);
    k_t2bf<<<(256 * 64 + 255) / 256, 256, 0, stream>>>(fcW1, fcW1t, 256, 64);

    const int gemmBlocks = (mtiles + 7) / 8;
    const int scatBlocks = (EE * 32 + 255) / 256;

    // GCN layer 1
    k_gemm128<<<gemmBlocks, 256, 0, stream>>>(bufBF, W1t, xl, mtiles);
    k_selfloop<<<(NF + 255) / 256, 256, 0, stream>>>(xl, dinv, agg, NF);
    k_scatter<<<scatBlocks, 256, 0, stream>>>(src, dstp, dinv, xl, agg, EE);
    k_bias_relu_bf<<<(NF + 255) / 256, 256, 0, stream>>>(agg, b1, bufBF, NF);

    // GCN layer 2
    k_gemm128<<<gemmBlocks, 256, 0, stream>>>(bufBF, W2t, xl, mtiles);
    k_selfloop<<<(NF + 255) / 256, 256, 0, stream>>>(xl, dinv, agg, NF);
    k_scatter<<<scatBlocks, 256, 0, stream>>>(src, dstp, dinv, xl, agg, EE);
    k_bias_relu_bf<<<(NF + 255) / 256, 256, 0, stream>>>(agg, b2, bufBF, NF);

    // fused BiLSTM + FC head
    k_lstm_fc<<<(mtiles + 3) / 4, 128, 0, stream>>>(bufBF, WihFb, WihBb,
        bihF, bhhF, bihB, bhhB, fcW1t, fcb1, fcW2, fcb2, out, mtiles);
}
